// CausalSelfAttentionMLA_34694745817818
// MI455X (gfx1250) — matrix operand order repaired and checked by reading, not compiled, not run on hardware
//
#include <hip/hip_runtime.h>
#include <hip/hip_bf16.h>

typedef __attribute__((ext_vector_type(16))) _Float16 v16h;
typedef __attribute__((ext_vector_type(8)))  float    v8f;

#define Bsz 2
#define Tsz 2048
#define Csz 2048
#define NHsz 16
#define Lsz 128

#if defined(__gfx1250__) && __has_builtin(__builtin_amdgcn_tensor_load_to_lds)
#define USE_TDM 1
#else
#define USE_TDM 0
#endif

#if USE_TDM
typedef __attribute__((ext_vector_type(4))) unsigned int u32x4;
typedef __attribute__((ext_vector_type(8))) int i32x8;
typedef __attribute__((ext_vector_type(4))) int i32x4;
#endif

__device__ __forceinline__ v8f wmma16x16x32(v16h a, v16h b, v8f c) {
    return __builtin_amdgcn_wmma_f32_16x16x32_f16(false, a, false, b, (short)0, c,
                                                  false, false);
}

__device__ __forceinline__ float rowmax16(float v) {
    #pragma unroll
    for (int m = 1; m < 16; m <<= 1) v = fmaxf(v, __shfl_xor(v, m, 32));
    return v;
}
__device__ __forceinline__ float rowsum16(float v) {
    #pragma unroll
    for (int m = 1; m < 16; m <<= 1) v += __shfl_xor(v, m, 32);
    return v;
}

// TDM: DMA a 32x128 f16 tile (row stride 128 elems) into LDS with the D# pad
// feature adding 16B of padding per 256B row (-> LDS row stride 272B).
__device__ __forceinline__ void tdm_load_ktile(const _Float16* gsrc, void* ldsDst)
{
#if USE_TDM
    const unsigned long long gaddr = (unsigned long long)(uintptr_t)gsrc;
    const unsigned int lds_off = (unsigned int)(uintptr_t)ldsDst;
    u32x4 g0;
    g0[0] = 1u;                                // count=1, user descriptor
    g0[1] = lds_off;                           // lds_addr (bytes)
    g0[2] = (unsigned int)gaddr;               // global_addr[31:0]
    g0[3] = (unsigned int)((gaddr >> 32) & 0x1FFFFFFu) | (2u << 30); // type=2
    i32x8 g1;
    g1[0] = (int)((1u << 16)      // data_size = 2 bytes
                | (1u << 20)      // pad_enable
                | (5u << 22)      // pad_interval: 64 DWORDs (=256B of data)
                | (3u << 25));    // pad_amount: 4 DWORDs (=16B pad)
    g1[1] = (int)(128u << 16);    // tensor_dim0 = 128 (lo16)
    g1[2] = (int)(32u << 16);     // tensor_dim0 hi=0 | tensor_dim1 = 32
    g1[3] = (int)(128u << 16);    // tensor_dim1 hi=0 | tile_dim0 = 128
    g1[4] = (int)32u;             // tile_dim1 = 32 | tile_dim2 = 0
    g1[5] = (int)128u;            // tensor_dim0_stride = 128 (lo32)
    g1[6] = 0;
    g1[7] = 0;
    i32x4 gz = {};
#if __clang_major__ >= 23
    i32x8 gz8 = {};
    __builtin_amdgcn_tensor_load_to_lds(g0, g1, gz, gz, gz8, 0);
#else
    __builtin_amdgcn_tensor_load_to_lds(g0, g1, gz, gz, 0);
#endif
#endif
}

// Permutation of a 32-element K-group into ISA A-fragment order:
// reads at [hi*16+i] are then 16 contiguous halfs per lane.
__device__ __forceinline__ int apos32(int kk) {
    return (kk < 8) ? kk : (kk < 16) ? kk + 8 : (kk < 24) ? kk - 8 : kk;
}

// ---------------------------------------------------------------------------
// Generic tiled GEMM: Out[M,N] = A[M,Kd] * W[Kd,N] + bias[N]
// Double-buffered, software-pipelined. A tile stored in permuted A-fragment
// order; W tile stored transposed: all fragment reads are 2x ds_load_b128.
// PERM_OUT stores the f16 output with each 32-column group permuted into
// A-fragment order (used for q, consumed as WMMA-A by the attention kernel).
// ---------------------------------------------------------------------------
template<int A_HALF, int OUT_HALF, int PERM_OUT>
__global__ __launch_bounds__(256)
void gemm_bias_kernel(const void* __restrict__ Ap, const float* __restrict__ W,
                      const float* __restrict__ bias, void* __restrict__ Outp,
                      int M, int N, int Kd)
{
    __shared__ _Float16 As [2][64][40];    // permuted [row][apos]
    __shared__ _Float16 BsT[2][128][40];   // transposed [n][k]

    const int tid   = threadIdx.x;
    const int w     = tid >> 5;
    const int lane  = tid & 31;
    const int m16   = lane & 15;
    const int hi    = lane >> 4;
    const int waveM = w & 3;
    const int waveN = w >> 2;
    const long long blockN0 = (long long)blockIdx.x * 128;
    const long long blockM0 = (long long)blockIdx.y * 64;

    v8f zero = {};
    v8f acc[4];
    #pragma unroll
    for (int t = 0; t < 4; t++) acc[t] = zero;

    const int arow  = tid >> 2;         // 0..63, 4 threads per row
    const int acol0 = (tid & 3) * 8;    // 8 consecutive k each
    // permuted A destination start (8-runs map to 8-runs: 0->0,8->16,16->8,24->24)
    const int apos0 = ((acol0 & 8) << 1) | ((acol0 & 16) >> 1);
    const int brow  = tid >> 3;         // 0..31 (k), 8 threads per row
    const int bcol0 = (tid & 7) * 16;   // 16 n-columns each

    const int nk = Kd >> 5;
    float    aF[8];
    _Float16 aH[8];
    float    bF[16];

    // ---- prologue: fetch + stage tile 0 ----
    {
        const long long gr = blockM0 + arow;
        if (A_HALF) {
            const _Float16* A = (const _Float16*)Ap;
            #pragma unroll
            for (int e = 0; e < 8; e++) aH[e] = A[gr * Kd + acol0 + e];
        } else {
            const float* A = (const float*)Ap;
            #pragma unroll
            for (int e = 0; e < 8; e++) aF[e] = A[gr * Kd + acol0 + e];
        }
        const long long wr = brow;
        #pragma unroll
        for (int e = 0; e < 16; e++)
            bF[e] = W[wr * (long long)N + blockN0 + bcol0 + e];
        #pragma unroll
        for (int e = 0; e < 8; e++)
            As[0][arow][apos0 + e] = A_HALF ? aH[e] : (_Float16)aF[e];
        #pragma unroll
        for (int e = 0; e < 16; e++)
            BsT[0][bcol0 + e][brow] = (_Float16)bF[e];
    }

    for (int kt = 0; kt < nk; kt++) {
        const int cur = kt & 1;
        __syncthreads();                         // publish buf[cur]
        const bool more = (kt + 1) < nk;
        if (more) {                              // fetch next tile (overlaps WMMA)
            const int k0 = (kt + 1) << 5;
            const long long gr = blockM0 + arow;
            if (A_HALF) {
                const _Float16* A = (const _Float16*)Ap;
                #pragma unroll
                for (int e = 0; e < 8; e++) aH[e] = A[gr * Kd + k0 + acol0 + e];
            } else {
                const float* A = (const float*)Ap;
                #pragma unroll
                for (int e = 0; e < 8; e++) aF[e] = A[gr * Kd + k0 + acol0 + e];
            }
            const long long wr = k0 + brow;
            #pragma unroll
            for (int e = 0; e < 16; e++)
                bF[e] = W[wr * (long long)N + blockN0 + bcol0 + e];
        }

        // ---- compute on buf[cur]: all fragment reads contiguous ----
        v16h a;
        #pragma unroll
        for (int i = 0; i < 16; i++)
            a[i] = As[cur][waveM * 16 + m16][hi * 16 + i];
        #pragma unroll
        for (int t = 0; t < 4; t++) {
            v16h b;
            #pragma unroll
            for (int i = 0; i < 16; i++)
                b[i] = BsT[cur][waveN * 64 + t * 16 + m16][hi * 8 + i + (i >= 8 ? 8 : 0)];
            acc[t] = wmma16x16x32(a, b, acc[t]);
        }

        if (more) {                              // stage next tile
            const int nxt = cur ^ 1;
            #pragma unroll
            for (int e = 0; e < 8; e++)
                As[nxt][arow][apos0 + e] = A_HALF ? aH[e] : (_Float16)aF[e];
            #pragma unroll
            for (int e = 0; e < 16; e++)
                BsT[nxt][bcol0 + e][brow] = (_Float16)bF[e];
        }
    }

    // epilogue: C/D layout -> lanes 0-15 rows 0..7, lanes 16-31 rows 8..15
    const int rbase = hi * 8;
    #pragma unroll
    for (int t = 0; t < 4; t++) {
        const int colIn32 = ((t & 1) << 4) + m16;
        const int colp = PERM_OUT ? apos32(colIn32) : colIn32;
        const long long colBase = blockN0 + waveN * 64 + (t >> 1) * 32;
        const float bv = bias[colBase + colIn32];
        #pragma unroll
        for (int r = 0; r < 8; r++) {
            const long long row = blockM0 + waveM * 16 + rbase + r;
            const float v = acc[t][r] + bv;
            if (OUT_HALF)
                ((_Float16*)Outp)[row * (long long)N + colBase + colp] = (_Float16)v;
            else
                ((float*)Outp)[row * (long long)N + colBase + colp] = v;
        }
    }
}

// ---------------------------------------------------------------------------
// Flash attention over the shared latent (K == V == k_lat, f16).
// TDM double-buffers 32x128 key tiles into LDS (DMA overlapped with WMMA);
// a cooperative LDS transpose produces KsT so the O-accumulate B-fragments
// are contiguous. q is pre-permuted so Q A-fragments load as b128 pairs.
// ---------------------------------------------------------------------------
__global__ __launch_bounds__(256)
void mla_flash_attention_kernel(const _Float16* __restrict__ kbuf,  // [B*T, L]
                                const _Float16* __restrict__ qbuf,  // [B*T, NH*L] permuted
                                _Float16* __restrict__ ybuf)        // [B*T, NH*L]
{
    __shared__ _Float16 Ks[2][32][136];   // double-buffered key tiles (+pad)
    __shared__ _Float16 KsT[128][40];     // transposed current tile [l][key]
    __shared__ _Float16 Ps[8][16][40];    // per-wave P scratch, permuted order

    const int tid  = threadIdx.x;
    const int w    = tid >> 5;
    const int lane = tid & 31;
    const int m16  = lane & 15;
    const int hi   = lane >> 4;

    const int bid  = blockIdx.x;
    const int tile = bid & 15;            // T/128 = 16 row tiles
    const int h    = (bid >> 4) & 15;
    const int b    = bid >> 8;
    const int R0   = tile * 128 + w * 16; // first query row of this wave
    const int rbase = hi * 8;

    // Q block as 4 A-fragments; qbuf is permuted -> contiguous 16 halfs/lane
    v16h qf[4];
    {
        const long long qrow =
            ((long long)(b * Tsz + R0 + m16)) * (NHsz * Lsz) + h * Lsz;
        #pragma unroll
        for (int c = 0; c < 4; c++) {
            #pragma unroll
            for (int i = 0; i < 16; i++)
                qf[c][i] = qbuf[qrow + c * 32 + hi * 16 + i];
        }
    }

    v8f zero = {};
    v8f o[8];
    #pragma unroll
    for (int l = 0; l < 8; l++) o[l] = zero;
    float mrun[8], srun[8];
    #pragma unroll
    for (int r = 0; r < 8; r++) { mrun[r] = -1e30f; srun[r] = 0.0f; }

    const float scale = 0.08838834764831845f; // 1/sqrt(128)
    const int nsteps = (tile + 1) * 4;        // 32 keys per step, causal bound

#if USE_TDM
    if (w == 0)   // preload tile 0
        tdm_load_ktile(kbuf + ((long long)(b * Tsz)) * Lsz, (void*)&Ks[0][0][0]);
#endif

    for (int s = 0; s < nsteps; s++) {
        const int cur = s & 1;
        const int s0 = s * 32;
#if USE_TDM
        if (w == 0) __builtin_amdgcn_s_wait_tensorcnt(0);
        __syncthreads();                  // publish Ks[cur]
        if (w == 0 && s + 1 < nsteps)     // DMA next tile while we compute
            tdm_load_ktile(kbuf + ((long long)(b * Tsz + s0 + 32)) * Lsz,
                           (void*)&Ks[cur ^ 1][0][0]);
#else
        __syncthreads();
        {
            const int  krow  = tid >> 3;
            const int  kcol0 = (tid & 7) * 16;
            const long long gr = ((long long)(b * Tsz + s0 + krow)) * Lsz;
            #pragma unroll
            for (int e = 0; e < 16; e++)
                Ks[cur][krow][kcol0 + e] = kbuf[gr + kcol0 + e];
        }
        __syncthreads();
#endif
        // cooperative transpose Ks[cur] -> KsT (for O-accumulate B-fragments)
        {
            const int krow  = tid >> 3;
            const int kcol0 = (tid & 7) * 16;
            _Float16 tmp[16];
            #pragma unroll
            for (int e = 0; e < 16; e++) tmp[e] = Ks[cur][krow][kcol0 + e];
            #pragma unroll
            for (int e = 0; e < 16; e++) KsT[kcol0 + e][krow] = tmp[e];
        }
        __syncthreads();

        // S = Q * K^T  (contract over L=128 in 4 chunks), two 16-key groups
        v8f sacc[2];
        sacc[0] = zero; sacc[1] = zero;
        #pragma unroll
        for (int nt = 0; nt < 2; nt++) {
            #pragma unroll
            for (int c = 0; c < 4; c++) {
                v16h bf;   // B[kk][n] = Ktile[key = nt*16+n][l = c*32+kk]
                #pragma unroll
                for (int i = 0; i < 16; i++)
                    bf[i] = Ks[cur][nt * 16 + m16][c * 32 + hi * 8 + i + (i >= 8 ? 8 : 0)];
                sacc[nt] = wmma16x16x32(qf[c], bf, sacc[nt]);
            }
        }

        // scale + causal mask (C layout: col = m16, row = rbase + r)
        #pragma unroll
        for (int nt = 0; nt < 2; nt++) {
            #pragma unroll
            for (int r = 0; r < 8; r++) {
                const int key = s0 + nt * 16 + m16;
                const int row = R0 + rbase + r;
                const float v = sacc[nt][r] * scale;
                sacc[nt][r] = (key > row) ? -1e30f : v;
            }
        }

        // online softmax; write P to per-wave LDS in permuted A-frag order
        const int pos0 = (m16 < 8) ? m16 : m16 + 8;        // key = m16
        const int pos1 = (m16 < 8) ? m16 + 8 : m16 + 16;   // key = 16+m16
        float alpha[8];
        #pragma unroll
        for (int r = 0; r < 8; r++) {
            float mx = fmaxf(sacc[0][r], sacc[1][r]);
            mx = rowmax16(mx);
            const float mnew = fmaxf(mrun[r], mx);
            alpha[r] = __expf(mrun[r] - mnew);
            const float p0 = __expf(sacc[0][r] - mnew);
            const float p1 = __expf(sacc[1][r] - mnew);
            Ps[w][rbase + r][pos0] = (_Float16)p0;
            Ps[w][rbase + r][pos1] = (_Float16)p1;
            srun[r] = srun[r] * alpha[r] + rowsum16(p0 + p1);
            mrun[r] = mnew;
        }
        // rescale existing accumulator
        #pragma unroll
        for (int l = 0; l < 8; l++) {
            #pragma unroll
            for (int r = 0; r < 8; r++) o[l][r] *= alpha[r];
        }

        // read P back as a 16x32 A-fragment (contiguous, same-wave LDS order)
        v16h pa;
        #pragma unroll
        for (int i = 0; i < 16; i++)
            pa[i] = Ps[w][m16][hi * 16 + i];

        // O += P * Ktile: B-fragments contiguous from KsT
        #pragma unroll
        for (int l = 0; l < 8; l++) {
            v16h bf;   // B[kk][n] = KsT[l*16 + n][key = kk]
            #pragma unroll
            for (int i = 0; i < 16; i++)
                bf[i] = KsT[l * 16 + m16][hi * 8 + i + (i >= 8 ? 8 : 0)];
            o[l] = wmma16x16x32(pa, bf, o[l]);
        }
    }

    // normalize and store y as f16 in [B,T,NH*L] layout (== [B,T,H,L] flat)
    #pragma unroll
    for (int r = 0; r < 8; r++) {
        const float inv = 1.0f / srun[r];
        const long long yrow =
            ((long long)(b * Tsz + R0 + rbase + r)) * (NHsz * Lsz) + h * Lsz;
        #pragma unroll
        for (int l = 0; l < 8; l++)
            ybuf[yrow + l * 16 + m16] = (_Float16)(o[l][r] * inv);
    }
}

// ---------------------------------------------------------------------------
extern "C" void kernel_launch(void* const* d_in, const int* in_sizes, int n_in,
                              void* d_out, int out_size, void* d_ws, size_t ws_size,
                              hipStream_t stream) {
    const float* x      = (const float*)d_in[0];
    const float* W_lat  = (const float*)d_in[1];
    const float* b_lat  = (const float*)d_in[2];
    const float* W_d    = (const float*)d_in[3];
    const float* b_d    = (const float*)d_in[4];
    const float* W_proj = (const float*)d_in[5];
    const float* b_proj = (const float*)d_in[6];
    float* out = (float*)d_out;

    const int M  = Bsz * Tsz;        // 4096
    const int QN = NHsz * Lsz;       // 2048

    // f16 workspace: k [M,L], q [M,NH*L] (permuted), y [M,NH*L]  (~33 MB)
    _Float16* kbuf = (_Float16*)d_ws;
    _Float16* qbuf = kbuf + (size_t)M * Lsz;
    _Float16* ybuf = qbuf + (size_t)M * QN;

    dim3 blk(256);

    // 1) k = x @ W_lat + b_lat  -> f16 (natural layout; TDM streams it)
    gemm_bias_kernel<0, 1, 0><<<dim3(Lsz / 128, M / 64), blk, 0, stream>>>(
        (const void*)x, W_lat, b_lat, (void*)kbuf, M, Lsz, Csz);

    // 2) q = x @ W_d + b_d  -> f16, 32-col groups permuted to A-frag order
    gemm_bias_kernel<0, 1, 1><<<dim3(QN / 128, M / 64), blk, 0, stream>>>(
        (const void*)x, W_d, b_d, (void*)qbuf, M, QN, Csz);

    // 3) flash attention over shared latent -> y (f16)
    mla_flash_attention_kernel<<<dim3(Bsz * NHsz * (Tsz / 128)), blk, 0, stream>>>(
        kbuf, qbuf, ybuf);

    // 4) out = y @ W_proj + b_proj  -> f32
    gemm_bias_kernel<1, 0, 0><<<dim3(Csz / 128, M / 64), blk, 0, stream>>>(
        (const void*)ybuf, W_proj, b_proj, (void*)out, M, Csz, QN);
}
